// PositionAwareAttention_2216203125494
// MI455X (gfx1250) — matrix operand order repaired and checked by reading, not compiled, not run on hardware
//
#include <hip/hip_runtime.h>
#include <hip/hip_bf16.h>
#include <math.h>

#define B_   64
#define S_   2048
#define DIN  512
#define DQ   512
#define DF   128
#define DA   256

#define NKS_X 16      // K-steps of 32 for x@Uw^T (K=512)
#define NKS_F 4       // K-steps of 32 for f@Ww^T (K=128)
#define NKS_H 10      // K-steps staged per LDS phase
#define ROWSTRIDE 12  // dwords per (row, khalf): 8 data + 4 pad -> conflict-free banks

typedef __attribute__((ext_vector_type(16))) __bf16    v16bf;
typedef __attribute__((ext_vector_type(8)))  float     v8f;
typedef __attribute__((ext_vector_type(4)))  unsigned  v4u;
typedef __attribute__((ext_vector_type(8)))  unsigned  v8u;

// two f32 -> packed bf16 dword (hardware cvt via __bf16 conversion)
__device__ __forceinline__ unsigned pack_bf16(float lo, float hi) {
  __bf16 l = (__bf16)lo;
  __bf16 h = (__bf16)hi;
  unsigned ul = (unsigned)__builtin_bit_cast(unsigned short, l);
  unsigned uh = (unsigned)__builtin_bit_cast(unsigned short, h);
  return ul | (uh << 16);
}

__device__ __forceinline__ v16bf frag_from_dwords(const unsigned* p) {
  v4u lo = *reinterpret_cast<const v4u*>(p);
  v4u hi = *reinterpret_cast<const v4u*>(p + 4);
  v8u v  = __builtin_shufflevector(lo, hi, 0, 1, 2, 3, 4, 5, 6, 7);
  return __builtin_bit_cast(v16bf, v);
}

// tanh via hardware exp/rcp: correct limits (+/-1) for large |v|
__device__ __forceinline__ float fast_tanh(float v) {
  const float e = __expf(2.0f * v);     // v_exp_f32
  return 1.0f - 2.0f / (e + 1.0f);      // v_rcp_f32
}

// ---------------------------------------------------------------------------
// Weight pre-pack: W [NA][K] f32  ->  fragment-ordered bf16.
// Fragment (nt, ks) = 32x16 B tile: lane holds column n = lane&15,
// elements e=0..15 hold K = ks*32 + 16*(lane>>4) + e  (contiguous in W).
// Dest dword index: (frag*32 + lane)*8 + j,  frag = nt*(K/32) + ks.
// ---------------------------------------------------------------------------
template <int K>
__global__ __launch_bounds__(256) void paa_packw_kernel(
    const float* __restrict__ W, unsigned* __restrict__ P) {
  constexpr int NKS = K / 32;
  const int tid  = blockIdx.x * 256 + threadIdx.x;
  const int lane = tid & 31;
  const int frag = tid >> 5;
  const int nt   = frag / NKS;
  const int ks   = frag % NKS;
  const int n    = lane & 15;
  const int kh   = lane >> 4;
  const float* src = W + (size_t)(nt * 16 + n) * K + ks * 32 + kh * 16;
  unsigned* dst = P + ((size_t)frag * 32 + lane) * 8;
#pragma unroll
  for (int j = 0; j < 8; ++j) dst[j] = pack_bf16(src[2 * j], src[2 * j + 1]);
}

// ---------------------------------------------------------------------------
// Kernel 1: qb[b,a] = Ub[a] + sum_k q[b,k] * Vw[a,k]     (tiny)
// ---------------------------------------------------------------------------
__global__ __launch_bounds__(DA) void paa_qproj_kernel(
    const float* __restrict__ q, const float* __restrict__ Vw,
    const float* __restrict__ Ub, float* __restrict__ qb) {
  const int b = blockIdx.x;
  const int a = threadIdx.x;
  const float* qr = q + (size_t)b * DQ;
  const float* vr = Vw + (size_t)a * DQ;
  float acc = Ub[a];
#pragma unroll 4
  for (int k = 0; k < DQ; k += 4) {
    acc += qr[k] * vr[k] + qr[k+1] * vr[k+1] + qr[k+2] * vr[k+2] + qr[k+3] * vr[k+3];
  }
  qb[(size_t)b * DA + a] = acc;
}

// ---------------------------------------------------------------------------
// A-tile staging: convert NSTEP K-steps (32 K each) of a 64-row f32 tile to
// fragment-ordered bf16 in LDS. Thread owns fixed (r = tid>>4, p = tid&15);
// all loop strides are compile-time constants -> immediate-offset loads/stores.
// ---------------------------------------------------------------------------
template <int NSTEP>
__device__ __forceinline__ void stage_tile(
    const float* __restrict__ src_base, int ld, int k0f, int slot0,
    unsigned* __restrict__ s_a, int r, int p) {
  const int wkh = p >> 3;                       // contiguous K layout: k = 16*kh + e
  const int wj  = p & 7;
#pragma unroll
  for (int rg = 0; rg < 4; ++rg) {
    const int row = rg * 16 + r;
    const float* src = src_base + (size_t)row * ld + k0f + 2 * p;
    unsigned* dst = &s_a[((slot0 * 2 + wkh) * 64 + row) * ROWSTRIDE + wj];
#pragma unroll
    for (int l = 0; l < NSTEP; ++l) {
      const float2 t = *reinterpret_cast<const float2*>(src + l * 32);
      dst[l * (2 * 64 * ROWSTRIDE)] = pack_bf16(t.x, t.y);
    }
  }
}

// ---------------------------------------------------------------------------
// NSTEP K-steps of WMMA: A frags from LDS (imm-offset ds_load_b128),
// B frags from pre-packed global (saddr + lane*32 + imm offset).
// ---------------------------------------------------------------------------
template <int NSTEP, int NKS>
__device__ __forceinline__ void compute_steps(
    const unsigned* __restrict__ s_a, int slot0,
    const unsigned* __restrict__ BP, int ks0,
    int wave, int lane, int kh, int r16, v8f (&acc)[4][2]) {
#pragma unroll
  for (int l = 0; l < NSTEP; ++l) {
    v16bf afr[4];
#pragma unroll
    for (int mt = 0; mt < 4; ++mt) {
      const int base = (((slot0 + l) * 2 + kh) * 64 + mt * 16 + r16) * ROWSTRIDE;
      afr[mt] = frag_from_dwords(&s_a[base]);
    }
#pragma unroll
    for (int nt = 0; nt < 2; ++nt) {
      const unsigned* bp =
          BP + ((size_t)((wave * 2 + nt) * NKS + ks0 + l) * 32 + lane) * 8;
      const v16bf bfr = frag_from_dwords(bp);
#pragma unroll
      for (int mt = 0; mt < 4; ++mt)
        acc[mt][nt] = __builtin_amdgcn_wmma_f32_16x16x32_bf16(
            false, afr[mt], false, bfr, (short)0, acc[mt][nt], false, false);
    }
  }
}

// ---------------------------------------------------------------------------
// Kernel 2: scores[b,s] = Tw . tanh(x@Uw^T + qb + f@Ww^T) + Tb   (WMMA bf16)
// Block = 64 s-rows of one batch; 8 waves x (4 M-tiles x 2 N-tiles).
// ---------------------------------------------------------------------------
__global__ __launch_bounds__(256) void paa_scores_kernel(
    const float* __restrict__ x, const float* __restrict__ f,
    const unsigned* __restrict__ UwP, const unsigned* __restrict__ WwP,
    const float* __restrict__ qb, const float* __restrict__ Tw,
    const float* __restrict__ Tb, const unsigned char* __restrict__ mask,
    float* __restrict__ scores) {
  const int b    = blockIdx.y;
  const int s0   = blockIdx.x * 64;
  const int wave = threadIdx.x >> 5;
  const int lane = threadIdx.x & 31;
  const int r16  = lane & 15;
  const int kh   = lane >> 4;
  const int sr   = threadIdx.x >> 4;   // staging row 0..15
  const int sp   = threadIdx.x & 15;   // staging K-pair 0..15

  // A staging: [slot (10)][kh (2)][row (64)][ROWSTRIDE dwords]  = 60 KB
  __shared__ unsigned s_a[NKS_H * 2 * 64 * ROWSTRIDE];
  __shared__ float s_scores[64];
  if (threadIdx.x < 64) s_scores[threadIdx.x] = 0.0f;

  v8f acc[4][2];
#pragma unroll
  for (int mt = 0; mt < 4; ++mt)
#pragma unroll
    for (int nt = 0; nt < 2; ++nt) acc[mt][nt] = {};

  const float* xb = x + ((size_t)b * S_ + s0) * DIN;
  const float* fb = f + ((size_t)b * S_ + s0) * DF;

  // ---- phase 0: x K-steps 0..9 ----
  stage_tile<NKS_H>(xb, DIN, 0, 0, s_a, sr, sp);
  __syncthreads();
  compute_steps<NKS_H, NKS_X>(s_a, 0, UwP, 0, wave, lane, kh, r16, acc);
  __syncthreads();
  // ---- phase 1: x K-steps 10..15 (slots 0..5) + f K-steps 0..3 (slots 6..9)
  stage_tile<NKS_X - NKS_H>(xb, DIN, NKS_H * 32, 0, s_a, sr, sp);
  stage_tile<NKS_F>(fb, DF, 0, NKS_X - NKS_H, s_a, sr, sp);
  __syncthreads();
  compute_steps<NKS_X - NKS_H, NKS_X>(s_a, 0, UwP, NKS_H, wave, lane, kh, r16, acc);
  compute_steps<NKS_F, NKS_F>(s_a, NKS_X - NKS_H, WwP, 0, wave, lane, kh, r16, acc);

  // ---- epilogue: h = tanh(acc + qb[a]); reduce h*Tw[a] over a into LDS ----
  // C/D layout: VGPR g -> row = mt*16 + g + 8*kh, col a = ntg*16 + r16
  float qv[2], tw[2];
#pragma unroll
  for (int nt = 0; nt < 2; ++nt) {
    const int a = (wave * 2 + nt) * 16 + r16;
    qv[nt] = qb[(size_t)b * DA + a];
    tw[nt] = Tw[a];
  }
#pragma unroll
  for (int mt = 0; mt < 4; ++mt) {
#pragma unroll
    for (int g = 0; g < 8; ++g) {
      const float c = fast_tanh(acc[mt][0][g] + qv[0]) * tw[0]
                    + fast_tanh(acc[mt][1][g] + qv[1]) * tw[1];
      atomicAdd(&s_scores[mt * 16 + g + 8 * kh], c);  // ds_add_f32
    }
  }
  __syncthreads();

  if (threadIdx.x < 64) {
    const size_t s = (size_t)b * S_ + (s0 + threadIdx.x);
    const float sc = s_scores[threadIdx.x] + Tb[0];
    scores[s] = mask[s] ? -INFINITY : sc;
  }
}

// ---------------------------------------------------------------------------
// Kernel 3: in-place softmax over S per batch row
// ---------------------------------------------------------------------------
__global__ __launch_bounds__(256) void paa_softmax_kernel(float* __restrict__ scores) {
  const int b = blockIdx.x;
  const int t = threadIdx.x;
  __shared__ float red[256];
  float* row = scores + (size_t)b * S_;

  float m = -INFINITY;
  for (int s = t; s < S_; s += 256) m = fmaxf(m, row[s]);
  red[t] = m;
  __syncthreads();
  for (int o = 128; o > 0; o >>= 1) {
    if (t < o) red[t] = fmaxf(red[t], red[t + o]);
    __syncthreads();
  }
  m = red[0];
  __syncthreads();

  float sum = 0.0f;
  for (int s = t; s < S_; s += 256) {
    const float e = expf(row[s] - m);
    row[s] = e;
    sum += e;
  }
  red[t] = sum;
  __syncthreads();
  for (int o = 128; o > 0; o >>= 1) {
    if (t < o) red[t] += red[t + o];
    __syncthreads();
  }
  const float inv = 1.0f / red[0];
  for (int s = t; s < S_; s += 256) row[s] *= inv;
}

// ---------------------------------------------------------------------------
// Kernel 4: out[b,d] = sum_s w[b,s] * x[b,s,d]   (bandwidth-bound stream)
// ---------------------------------------------------------------------------
__global__ __launch_bounds__(128) void paa_out_kernel(
    const float* __restrict__ x, const float* __restrict__ w,
    float* __restrict__ out) {
  const int b = blockIdx.y;
  const int d = blockIdx.x * 128 + threadIdx.x;
  const float* xb = x + (size_t)b * S_ * DIN + d;
  const float* wb = w + (size_t)b * S_;
  float acc = 0.0f;
  for (int s = 0; s < S_; s += 4) {
    const float w0 = wb[s], w1 = wb[s + 1], w2 = wb[s + 2], w3 = wb[s + 3];
    acc += w0 * xb[(size_t)(s + 0) * DIN]
         + w1 * xb[(size_t)(s + 1) * DIN]
         + w2 * xb[(size_t)(s + 2) * DIN]
         + w3 * xb[(size_t)(s + 3) * DIN];
  }
  out[(size_t)b * DIN + d] = acc;
}

// ---------------------------------------------------------------------------
extern "C" void kernel_launch(void* const* d_in, const int* in_sizes, int n_in,
                              void* d_out, int out_size, void* d_ws, size_t ws_size,
                              hipStream_t stream) {
  const float*         x  = (const float*)d_in[0];
  const unsigned char* xm = (const unsigned char*)d_in[1];
  const float*         q  = (const float*)d_in[2];
  const float*         f  = (const float*)d_in[3];
  const float*         Uw = (const float*)d_in[4];
  const float*         Ub = (const float*)d_in[5];
  const float*         Vw = (const float*)d_in[6];
  const float*         Ww = (const float*)d_in[7];
  const float*         Tw = (const float*)d_in[8];
  const float*         Tb = (const float*)d_in[9];
  float* out = (float*)d_out;

  float*    ws     = (float*)d_ws;
  float*    qb     = ws;                       // B*DA floats
  float*    scores = ws + B_ * DA;             // B*S floats (weights in-place)
  unsigned* UwP    = (unsigned*)(scores + (size_t)B_ * S_);  // 256 frags * 256 dw
  unsigned* WwP    = UwP + 16 * NKS_X * 32 * 8;              // 64 frags * 256 dw

  paa_packw_kernel<DIN><<<dim3(32), dim3(256), 0, stream>>>(Uw, UwP);
  paa_packw_kernel<DF><<<dim3(8), dim3(256), 0, stream>>>(Ww, WwP);
  paa_qproj_kernel<<<dim3(B_), dim3(DA), 0, stream>>>(q, Vw, Ub, qb);
  paa_scores_kernel<<<dim3(S_ / 64, B_), dim3(256), 0, stream>>>(
      x, f, UwP, WwP, qb, Tw, Tb, xm, scores);
  paa_softmax_kernel<<<dim3(B_), dim3(256), 0, stream>>>(scores);
  paa_out_kernel<<<dim3(DIN / 128, B_), dim3(128), 0, stream>>>(x, scores, out);
}
